// VectorQuantize_59330678227554
// MI455X (gfx1250) — matrix operand order repaired and checked by reading, not compiled, not run on hardware
//
#include <hip/hip_runtime.h>

typedef __attribute__((ext_vector_type(16))) __bf16 v16bf;
typedef __attribute__((ext_vector_type(8)))  float  v8f;

namespace {
constexpr int kDim     = 256;     // embedding dim
constexpr int kCodes   = 8192;    // codebook size
constexpr int kRows    = 16384;   // 8 * 2048 flattened tokens
constexpr int kMT      = 64;      // rows per workgroup (4 waves x 16)
constexpr int kNT      = 32;      // codes staged to LDS per iteration
constexpr int kBlock   = 128;     // 4 waves (wave32)
constexpr int kQElems  = kRows * kDim;
constexpr int kIdxOff  = kQElems;            // indices after quantized
constexpr int kLossOff = kQElems + kRows;    // scalar loss last
constexpr float kLossScale = 1.25f / (float)kQElems;  // (1 + 0.25) * mean
}

__global__ void vq_zero_loss(float* out) { out[kLossOff] = 0.0f; }

__global__ __launch_bounds__(kBlock)
void vq_kernel(const float* __restrict__ x,
               const float* __restrict__ emb,
               float* __restrict__ out) {
  // Codebook tile staged as pre-split bf16 hi/lo (so B fragments are plain
  // contiguous 32B LDS reads in contiguous k order, matching A) + per-thread ||e||^2 partials.
  __shared__ __attribute__((aligned(32))) __bf16 s_ehi[kNT][kDim];
  __shared__ __attribute__((aligned(32))) __bf16 s_elo[kNT][kDim];
  __shared__ __attribute__((aligned(16))) float  s_part[kBlock];
  __shared__ int   s_idx[kMT];
  __shared__ float s_wsum[kBlock / 32];

  const int t    = threadIdx.x;
  const int wave = t >> 5;
  const int lane = t & 31;
  const int half = lane >> 4;   // which 16-lane half of the wave
  const int l16  = lane & 15;
  const int rowbase = blockIdx.x * kMT;
  const int myrow   = rowbase + wave * 16 + l16;

  // ---- Preload this wave's 16xK A tile as bf16 hi/lo WMMA fragments ------
  // 16-bit A 16x32 layout: lanes 0-15 row M=l16 hold K {kc..kc+15};
  // lanes 16-31 hold K {kc+16..kc+31} (contiguous, matching B).
  v16bf a_hi[8], a_lo[8];
  {
    const float* xr = x + (size_t)myrow * kDim;
#pragma unroll
    for (int j = 0; j < 8; ++j) {
      const int kc = j * 32;
      const int o0 = kc + half * 16;
      const int o1 = kc + half * 16 + 8;
#pragma unroll
      for (int i = 0; i < 8; ++i) {
        float f0 = xr[o0 + i];
        __bf16 h0 = (__bf16)f0;
        a_hi[j][i]     = h0;
        a_lo[j][i]     = (__bf16)(f0 - (float)h0);
        float f1 = xr[o1 + i];
        __bf16 h1 = (__bf16)f1;
        a_hi[j][8 + i] = h1;
        a_lo[j][8 + i] = (__bf16)(f1 - (float)h1);
      }
    }
  }

  float minval[8];
  int   minidx[8];
#pragma unroll
  for (int r = 0; r < 8; ++r) { minval[r] = 3.0e38f; minidx[r] = 0; }

  const int sc = t >> 2;          // staged code within tile (0..31)
  const int sk = (t & 3) * 64;    // staged k-base (64 floats per thread)

  for (int n0 = 0; n0 < kCodes; n0 += kNT) {
    __syncthreads();              // previous iteration's LDS consumers done

    // ---- stage 32 codes: global f32 -> LDS bf16 hi/lo, plus ||e||^2 ----
    const float* gsrc = emb + (size_t)(n0 + sc) * kDim + sk;
    if (n0 + kNT < kCodes)
      __builtin_prefetch(gsrc + kNT * kDim, 0, 1);   // global_prefetch_b8
    float psum = 0.0f;
#pragma unroll
    for (int i = 0; i < 64; i += 4) {
      float4 v = *(const float4*)(gsrc + i);
      float fv0 = v.x, fv1 = v.y, fv2 = v.z, fv3 = v.w;
      __bf16 h0 = (__bf16)fv0, h1 = (__bf16)fv1;
      __bf16 h2 = (__bf16)fv2, h3 = (__bf16)fv3;
      s_ehi[sc][sk + i + 0] = h0;  s_elo[sc][sk + i + 0] = (__bf16)(fv0 - (float)h0);
      s_ehi[sc][sk + i + 1] = h1;  s_elo[sc][sk + i + 1] = (__bf16)(fv1 - (float)h1);
      s_ehi[sc][sk + i + 2] = h2;  s_elo[sc][sk + i + 2] = (__bf16)(fv2 - (float)h2);
      s_ehi[sc][sk + i + 3] = h3;  s_elo[sc][sk + i + 3] = (__bf16)(fv3 - (float)h3);
      psum += fv0 * fv0 + fv1 * fv1 + fv2 * fv2 + fv3 * fv3;
    }
    s_part[t] = psum;
    __syncthreads();

    // ---- two 16-code sub-tiles: 8 k-steps x 3 bf16 WMMAs, 3 independent
    // accumulator chains per sub-tile so the XDL pipe never stalls on a
    // D->C dependency (6 interleavable chains in the unrolled body). ----
#pragma unroll
    for (int sub = 0; sub < 2; ++sub) {
      v8f c0 = {}, c1 = {}, c2 = {};
      const int code = sub * 16 + l16;   // B layout: lane n <-> column n
#pragma unroll
      for (int j = 0; j < 8; ++j) {
        const v16bf bhi = *(const v16bf*)&s_ehi[code][j * 32 + half * 16];
        const v16bf blo = *(const v16bf*)&s_elo[code][j * 32 + half * 16];
        c0 = __builtin_amdgcn_wmma_f32_16x16x32_bf16(false, a_hi[j], false, bhi,
                                                     (short)0, c0, false, false);
        c1 = __builtin_amdgcn_wmma_f32_16x16x32_bf16(false, a_lo[j], false, bhi,
                                                     (short)0, c1, false, false);
        c2 = __builtin_amdgcn_wmma_f32_16x16x32_bf16(false, a_hi[j], false, blo,
                                                     (short)0, c2, false, false);
      }
      // ||e_code||^2: sum of this code's 4 staged partials (one b128 read).
      const float4 p = *(const float4*)&s_part[code * 4];
      const float  en = (p.x + p.y) + (p.z + p.w);
      const int    n  = n0 + code;
#pragma unroll
      for (int r = 0; r < 8; ++r) {
        const float dot  = c0[r] + c1[r] + c2[r];
        const float dist = en - 2.0f * dot;     // ||x||^2 dropped (row-const)
        if (dist < minval[r]) { minval[r] = dist; minidx[r] = n; }
      }
    }
  }

  // ---- argmin across the 16 lanes of each C half (lexicographic ties) ----
#pragma unroll
  for (int mask = 1; mask <= 8; mask <<= 1) {
#pragma unroll
    for (int r = 0; r < 8; ++r) {
      const float ov = __shfl_xor(minval[r], mask, 32);
      const int   oi = __shfl_xor(minidx[r], mask, 32);
      if (ov < minval[r] || (ov == minval[r] && oi < minidx[r])) {
        minval[r] = ov; minidx[r] = oi;
      }
    }
  }
  if (l16 == 0) {                 // lane 0 -> rows r, lane 16 -> rows r+8
#pragma unroll
    for (int r = 0; r < 8; ++r) {
      const int m = wave * 16 + half * 8 + r;
      s_idx[m] = minidx[r];
      out[kIdxOff + rowbase + m] = (float)minidx[r];
    }
  }
  __syncthreads();

  // ---- gather quantized rows (coalesced) + loss ---------------------------
  float lsum = 0.0f;
  for (int e = t; e < kMT * kDim; e += kBlock) {
    const int m  = e >> 8;        // row within tile
    const int k  = e & 255;
    const int ci = s_idx[m];
    const float q = emb[(size_t)ci * kDim + k];
    const int row = rowbase + m;
    out[(size_t)row * kDim + k] = q;
    const float d = x[(size_t)row * kDim + k] - q;
    lsum += d * d;
  }
#pragma unroll
  for (int mask = 1; mask <= 16; mask <<= 1) lsum += __shfl_xor(lsum, mask, 32);
  if (lane == 0) s_wsum[wave] = lsum;
  __syncthreads();
  if (t == 0) {
    float s = 0.0f;
#pragma unroll
    for (int w = 0; w < kBlock / 32; ++w) s += s_wsum[w];
    atomicAdd(out + kLossOff, s * kLossScale);
  }
}

extern "C" void kernel_launch(void* const* d_in, const int* in_sizes, int n_in,
                              void* d_out, int out_size, void* d_ws, size_t ws_size,
                              hipStream_t stream) {
  const float* x   = (const float*)d_in[0];
  const float* emb = (const float*)d_in[1];
  float* out = (float*)d_out;
  vq_zero_loss<<<1, 1, 0, stream>>>(out);
  vq_kernel<<<dim3(kRows / kMT), dim3(kBlock), 0, stream>>>(x, emb, out);
}
